// SpatialAwareGATConv_89481348645083
// MI455X (gfx1250) — matrix operand order repaired and checked by reading, not compiled, not run on hardware
//
#include <hip/hip_runtime.h>
#include <math.h>

#define NN 50000
#define EE 800000
#define INCH 256
#define EDGE_DIM 32
#define HH 4
#define HC 128  // H*C

typedef __attribute__((ext_vector_type(16))) _Float16 v16h;
typedef __attribute__((ext_vector_type(4)))  _Float16 v4h;
typedef __attribute__((ext_vector_type(8)))  float    v8f;

union B16 { v16h v; uint4 q[2]; };

__device__ __forceinline__ unsigned f2ord(float f) {
    unsigned u = __float_as_uint(f);
    return (u & 0x80000000u) ? ~u : (u | 0x80000000u);
}
__device__ __forceinline__ float ord2f(unsigned u) {
    return __uint_as_float((u & 0x80000000u) ? (u & 0x7fffffffu) : ~u);
}

__device__ __forceinline__ v4h cvt4(float4 f) {
    v4h h;
    h[0] = (_Float16)f.x; h[1] = (_Float16)f.y;
    h[2] = (_Float16)f.z; h[3] = (_Float16)f.w;
    return h;
}

// Build 16-bit A fragment (16x32, ISA interleaved K layout) from 4 float4 runs.
__device__ __forceinline__ v16h make_a_frag(const float* __restrict__ rowp, int half) {
    float4 f0 = *(const float4*)(rowp + half * 8);
    float4 f1 = *(const float4*)(rowp + half * 8 + 4);
    float4 f2 = *(const float4*)(rowp + 16 + half * 8);
    float4 f3 = *(const float4*)(rowp + 16 + half * 8 + 4);
    v16h a;
    a[0]=(_Float16)f0.x; a[1]=(_Float16)f0.y; a[2]=(_Float16)f0.z; a[3]=(_Float16)f0.w;
    a[4]=(_Float16)f1.x; a[5]=(_Float16)f1.y; a[6]=(_Float16)f1.z; a[7]=(_Float16)f1.w;
    a[8]=(_Float16)f2.x; a[9]=(_Float16)f2.y; a[10]=(_Float16)f2.z; a[11]=(_Float16)f2.w;
    a[12]=(_Float16)f3.x; a[13]=(_Float16)f3.y; a[14]=(_Float16)f3.z; a[15]=(_Float16)f3.w;
    return a;
}

// ---------------- K0: init accumulators ----------------
__global__ void init_kernel(float* __restrict__ out, unsigned* __restrict__ mord,
                            float* __restrict__ s) {
    int tid = blockIdx.x * 256 + threadIdx.x;
    if (tid < NN * HC) out[tid] = 0.0f;
    if (tid < NN * HH) { mord[tid] = 0u; s[tid] = 0.0f; }
}

// ---------------- K1: x@W_src and x@W_dst via WMMA f16 ----------------
__global__ void __launch_bounds__(256)
node_gemm_kernel(const float* __restrict__ x, const float* __restrict__ Wsrc,
                 const float* __restrict__ Wdst,
                 float* __restrict__ xs, float* __restrict__ xd) {
    __shared__ _Float16 ldsB[2][32 * 136];  // K-panel 32 x 128, padded rows (16B-aligned)
    const int t = threadIdx.x;
    const int wave = t >> 5, lane = t & 31;
    const int r = lane & 15, half = lane >> 4;
    int tile = blockIdx.x * 8 + wave;
    const bool active = tile < (NN / 16);
    if (!active) tile = 0;
    const int row_base = tile * 16;
    const int row = row_base + r;

    // staging addresses: thread owns 16 consecutive elements per matrix per panel
    const int sbase = t * 16;               // 0..4095, stays inside one K row
    const int sk = sbase >> 7, sn = sbase & 127;
    _Float16* q0 = &ldsB[0][sk * 136 + sn];
    _Float16* q1 = &ldsB[1][sk * 136 + sn];

    v8f zero = {};
    v8f acc[2][8];
#pragma unroll
    for (int m = 0; m < 2; ++m)
#pragma unroll
        for (int nt = 0; nt < 8; ++nt) acc[m][nt] = zero;

    for (int kk = 0; kk < 8; ++kk) {
        // stage both W panels: float4 loads -> packed f16 ds_store_b64
        const float4* ps = (const float4*)(Wsrc + kk * 4096 + sbase);
        const float4* pd = (const float4*)(Wdst + kk * 4096 + sbase);
#pragma unroll
        for (int i = 0; i < 4; ++i) {
            *(v4h*)(q0 + 4 * i) = cvt4(ps[i]);
            *(v4h*)(q1 + 4 * i) = cvt4(pd[i]);
        }
        __syncthreads();

        v16h a = make_a_frag(x + row * INCH + kk * 32, half);

#pragma unroll
        for (int m = 0; m < 2; ++m) {
            const _Float16* brow = &ldsB[m][r];
#pragma unroll
            for (int nt = 0; nt < 8; ++nt) {
                B16 b;
#pragma unroll
                for (int j = 0; j < 16; ++j)
                    b.v[j] = brow[((j & 8) * 2 + half * 8 + (j & 7)) * 136 + nt * 16];
                acc[m][nt] = __builtin_amdgcn_wmma_f32_16x16x32_f16(
                    false, a, false, b.v, (short)0, acc[m][nt], false, false);
            }
        }
        __syncthreads();
    }

    if (active) {
#pragma unroll
        for (int nt = 0; nt < 8; ++nt)
#pragma unroll
            for (int v = 0; v < 8; ++v) {
                int rr = row_base + half * 8 + v;
                xs[rr * HC + nt * 16 + r] = acc[0][nt][v];
                xd[rr * HC + nt * 16 + r] = acc[1][nt][v];
            }
    }
}

// ---------------- K2: fused edge GEMM (WMMA) + attention logits ----------------
__global__ void __launch_bounds__(128)
edge_logits_kernel(const float* __restrict__ edge_attr, const float* __restrict__ Wedge,
                   const int* __restrict__ edge_index, const float* __restrict__ dist_weight,
                   const float* __restrict__ att, const float* __restrict__ sscale,
                   const float* __restrict__ xs, const float* __restrict__ xd,
                   float* __restrict__ alpha_raw_out, float* __restrict__ adec,
                   unsigned* __restrict__ mord) {
    __shared__ _Float16 ldsW[32 * 136];       // W_edge f16, padded rows
    __shared__ float    ldsEF[4][16 * 132];   // per-wave e_feat tile 16x128, padded
    const int t = threadIdx.x;
    const int wave = t >> 5, lane = t & 31;
    const int r = lane & 15, half = lane >> 4;

    // stage W_edge (32x128): 32 consecutive elements per thread (within one K row)
    {
        const int sbase = t * 32;             // 0..4095
        const int sk = sbase >> 7, sn = sbase & 127;
        const float4* ps = (const float4*)(Wedge + sbase);
        _Float16* q = &ldsW[sk * 136 + sn];
#pragma unroll
        for (int i = 0; i < 8; ++i)
            *(v4h*)(q + 4 * i) = cvt4(ps[i]);
    }
    __syncthreads();

    const int ebase = (blockIdx.x * 4 + wave) * 16;

    // A fragment: 16 edges x 32 edge-dims; single K-step
    v16h a = make_a_frag(edge_attr + (ebase + r) * EDGE_DIM, half);
    const _Float16* brow = &ldsW[r];
    float* ef = &ldsEF[wave][0];
    v8f zero = {};
#pragma unroll
    for (int nt = 0; nt < 8; ++nt) {
        B16 b;
#pragma unroll
        for (int j = 0; j < 16; ++j)
            b.v[j] = brow[((j & 8) * 2 + half * 8 + (j & 7)) * 136 + nt * 16];
        v8f c = __builtin_amdgcn_wmma_f32_16x16x32_f16(
            false, a, false, b.v, (short)0, zero, false, false);
#pragma unroll
        for (int v = 0; v < 8; ++v)
            ef[(half * 8 + v) * 132 + nt * 16 + r] = c[v];
    }
    __syncthreads();

    // lane -> 4 channels (c0..c0+3) of one head
    const int h = lane >> 3;
    const float4 attv = *(const float4*)(att + 4 * lane);
    const float ssc = sscale[h];

    for (int el = 0; el < 16; ++el) {
        int e = ebase + el;
        int src = edge_index[e];
        int dst = edge_index[EE + e];
        float4 xsv = *(const float4*)(xs + src * HC + 4 * lane);
        float4 xdv = *(const float4*)(xd + dst * HC + 4 * lane);
        float4 efv = *(const float4*)(ef + el * 132 + 4 * lane);
        float p = tanhf(xsv.x + xdv.x + efv.x) * attv.x
                + tanhf(xsv.y + xdv.y + efv.y) * attv.y
                + tanhf(xsv.z + xdv.z + efv.z) * attv.z
                + tanhf(xsv.w + xdv.w + efv.w) * attv.w;
        p += __shfl_xor(p, 1, 8);
        p += __shfl_xor(p, 2, 8);
        p += __shfl_xor(p, 4, 8);
        if ((lane & 7) == 0) {
            alpha_raw_out[e * HH + h] = p;
            float ad = p * powf(dist_weight[e], ssc);
            adec[e * HH + h] = ad;
            atomicMax(&mord[dst * HH + h], f2ord(ad));
        }
    }
}

// ---------------- K3: ea = exp(alpha - m[dst]); s += ea ----------------
__global__ void __launch_bounds__(256)
ea_kernel(const int* __restrict__ edge_index, const unsigned* __restrict__ mord,
          float* __restrict__ adec_ea, float* __restrict__ s) {
    int tid = blockIdx.x * 256 + threadIdx.x;   // < E*H
    int e = tid >> 2, h = tid & 3;
    int dst = edge_index[EE + e];
    float m = ord2f(mord[dst * HH + h]);
    float ea = expf(adec_ea[tid] - m);
    adec_ea[tid] = ea;
    atomicAdd(&s[dst * HH + h], ea);
}

// ---------------- K4: normalize + weighted scatter-add ----------------
__global__ void __launch_bounds__(256)
agg_kernel(const int* __restrict__ edge_index, const float* __restrict__ ea,
           const float* __restrict__ s, const float* __restrict__ xs,
           float* __restrict__ out, float* __restrict__ alpha_norm_out) {
    int t = threadIdx.x;
    int lane = t & 31;
    int e = blockIdx.x * 8 + (t >> 5);
    int h = lane >> 3;
    int src = edge_index[e];
    int dst = edge_index[EE + e];
    float an = ea[e * HH + h] / (s[dst * HH + h] + 1e-8f);
    if ((lane & 7) == 0) alpha_norm_out[e * HH + h] = an;
    float4 xv = *(const float4*)(xs + src * HC + 4 * lane);
    float* op = out + dst * HC + 4 * lane;
    atomicAdd(op + 0, xv.x * an);
    atomicAdd(op + 1, xv.y * an);
    atomicAdd(op + 2, xv.z * an);
    atomicAdd(op + 3, xv.w * an);
}

extern "C" void kernel_launch(void* const* d_in, const int* in_sizes, int n_in,
                              void* d_out, int out_size, void* d_ws, size_t ws_size,
                              hipStream_t stream) {
    (void)in_sizes; (void)n_in; (void)out_size; (void)ws_size;
    const float* x           = (const float*)d_in[0];
    const int*   edge_index  = (const int*)d_in[1];
    const float* edge_attr   = (const float*)d_in[2];
    const float* dist_weight = (const float*)d_in[3];
    const float* Wsrc        = (const float*)d_in[4];
    const float* Wdst        = (const float*)d_in[5];
    const float* Wedge       = (const float*)d_in[6];
    const float* att         = (const float*)d_in[7];
    const float* sscale      = (const float*)d_in[8];

    float* out        = (float*)d_out;                       // [N, 128]
    float* alpha_norm = out + (size_t)NN * HC;               // [E, 4]
    float* alpha_raw  = alpha_norm + (size_t)EE * HH;        // [E, 4]

    float* xs   = (float*)d_ws;                              // [N, 128]
    float* xd   = xs + (size_t)NN * HC;                      // [N, 128]
    float* adec = xd + (size_t)NN * HC;                      // [E, 4] -> becomes ea
    unsigned* mord = (unsigned*)(adec + (size_t)EE * HH);    // [N, 4]
    float* s    = (float*)(mord + (size_t)NN * HH);          // [N, 4]

    init_kernel<<<(NN * HC + 255) / 256, 256, 0, stream>>>(out, mord, s);
    node_gemm_kernel<<<(NN / 16 + 7) / 8, 256, 0, stream>>>(x, Wsrc, Wdst, xs, xd);
    edge_logits_kernel<<<EE / 64, 128, 0, stream>>>(edge_attr, Wedge, edge_index,
        dist_weight, att, sscale, xs, xd, alpha_raw, adec, mord);
    ea_kernel<<<(EE * HH) / 256, 256, 0, stream>>>(edge_index, mord, adec, s);
    agg_kernel<<<EE / 8, 256, 0, stream>>>(edge_index, adec, s, xs, out, alpha_norm);
}
